// ReferenceAttention_86406152061646
// MI455X (gfx1250) — matrix operand order repaired and checked by reading, not compiled, not run on hardware
//
#include <hip/hip_runtime.h>
#include <hip/hip_bf16.h>

#define HIDDEN   1024
#define HEADS    16
#define HEAD_DIM 64
#define SEQ      2048
#define BATCH    2
#define ROWS     (BATCH * SEQ)          // 4096
#define QK_SCALE 0.125f                 // 1/sqrt(64)

typedef __attribute__((ext_vector_type(16))) __bf16 v16bf;
typedef __attribute__((ext_vector_type(8)))  __bf16 v8bf;
typedef __attribute__((ext_vector_type(4)))  __bf16 v4bf;
typedef __attribute__((ext_vector_type(8)))  float  v8f;

static __device__ __forceinline__ v16bf join16(v8bf lo, v8bf hi) {
    return __builtin_shufflevector(lo, hi, 0,1,2,3,4,5,6,7,8,9,10,11,12,13,14,15);
}

// A-fragment (16x32 bf16, M x K). Lane = row; lanes 0-15 kbase=k0, lanes 16-31 kbase=k0+8.
// Holds K = kbase+{0..7} and kbase+16+{0..7}: two contiguous 16B runs.
static __device__ __forceinline__ v16bf load_a_frag(const __bf16* __restrict__ row, int kbase) {
    v8bf lo = *(const v8bf*)(row + kbase);
    v8bf hi = *(const v8bf*)(row + kbase + 16);
    return join16(lo, hi);
}

// B-fragment (32x16 bf16, K x N). Lane = column; lanes 0-15 kbase=k0, lanes 16-31 kbase=k0+8.
// Holds K = kbase+{0..7} and kbase+16+{0..7}: two contiguous 16B runs.
static __device__ __forceinline__ v16bf load_b_frag(const __bf16* __restrict__ col, int kbase) {
    v8bf lo = *(const v8bf*)(col + kbase);
    v8bf hi = *(const v8bf*)(col + kbase + 16);
    return join16(lo, hi);
}

static __device__ __forceinline__ v8f wmma_bf16(v16bf a, v16bf b, v8f c) {
    // v_wmma_f32_16x16x32_bf16
    return __builtin_amdgcn_wmma_f32_16x16x32_bf16(false, a, false, b, (short)0, c, false, false);
}

// Cross-half exchange (lane <-> lane^16) without touching the LDS pipe:
// v_permlanex16 (VALU) with identity selects; fallback to one shuffle.
static __device__ __forceinline__ float xhalf(float v) {
#if __has_builtin(__builtin_amdgcn_permlanex16)
    int vi = __builtin_bit_cast(int, v);
    int o  = __builtin_amdgcn_permlanex16(vi, vi, 0x76543210, 0xFEDCBA98, false, false);
    return __builtin_bit_cast(float, o);
#else
    return __shfl_xor(v, 16, 32);
#endif
}

// ---------------------------------------------------------------------------
// fp32 -> bf16 conversion (vectorized x4)
// ---------------------------------------------------------------------------
__global__ void cvt_f32_bf16(const float* __restrict__ src, __bf16* __restrict__ dst, int n) {
    int idx = (blockIdx.x * blockDim.x + threadIdx.x) * 4;
    if (idx < n) {
        float4 v = *(const float4*)(src + idx);
        v4bf o;
        o.x = (__bf16)v.x; o.y = (__bf16)v.y; o.z = (__bf16)v.z; o.w = (__bf16)v.w;
        *(v4bf*)(dst + idx) = o;
    }
}

// ---------------------------------------------------------------------------
// Transposed bf16 GEMM: D[mA, nB] = sum_k A[mA,k] * B[nB,k].
// A = weights (rows = output features -> C-layout 'r' dim, 8 consecutive per lane),
// B = activations (rows = tokens -> C-layout lane dim).
// Block: 128 threads = 4 waves in 2x2; wave tile 64x64; block tile 128x128.
// EPI==0: QKV epilogue -> Q (pre-scaled)/K packed b128 into [b,h,s,d]; V scatter to [b,h,d,s]
// EPI==1: fp32 packed rows into Cout[token*ldc + feature]
// ---------------------------------------------------------------------------
template <int EPI>
__global__ __launch_bounds__(128)
void gemm_bf16(const __bf16* __restrict__ A, const __bf16* __restrict__ B, int K, int ldc,
               float* __restrict__ Cout,
               __bf16* __restrict__ Qb, __bf16* __restrict__ Kb, __bf16* __restrict__ Vt) {
    const int lane = threadIdx.x & 31;
    const int wave = threadIdx.x >> 5;
    const int l16  = lane & 15;
    const int half = lane >> 4;
    const int m0 = blockIdx.y * 128 + (wave >> 1) * 64;   // feature dim
    const int n0 = blockIdx.x * 128 + (wave & 1) * 64;    // token dim

    const __bf16* arow[4];
    const __bf16* brow[4];
#pragma unroll
    for (int t = 0; t < 4; ++t) {
        arow[t] = A + (size_t)(m0 + t * 16 + l16) * K;
        brow[t] = B + (size_t)(n0 + t * 16 + l16) * K;
    }

    v8f acc[4][4] = {};

    for (int k0 = 0; k0 < K; k0 += 32) {
        if (k0 + 64 <= K) {  // global_prefetch_b8 next K-step
            __builtin_prefetch(arow[0] + k0 + 64, 0, 1);
            __builtin_prefetch(brow[0] + k0 + 64, 0, 1);
        }
        v16bf af[4], bf[4];
#pragma unroll
        for (int t = 0; t < 4; ++t) af[t] = load_a_frag(arow[t], k0 + half * 8);
#pragma unroll
        for (int t = 0; t < 4; ++t) bf[t] = load_b_frag(brow[t], k0 + half * 8);
#pragma unroll
        for (int mt = 0; mt < 4; ++mt)
#pragma unroll
            for (int nt = 0; nt < 4; ++nt)
                acc[mt][nt] = wmma_bf16(af[mt], bf[nt], acc[mt][nt]);
    }

    // C/D layout: element r @ lane L -> feature = m0+mt*16 + r + (L>=16?8:0), token = n0+nt*16+L%16
#pragma unroll
    for (int mt = 0; mt < 4; ++mt) {
        const int fbase = m0 + mt * 16 + half * 8;   // 8 consecutive features per lane
#pragma unroll
        for (int nt = 0; nt < 4; ++nt) {
            const int tok = n0 + nt * 16 + l16;
            if (EPI == 1) {
                *(v8f*)&Cout[(size_t)tok * ldc + fbase] = acc[mt][nt];   // 32B run
            } else {
                const int bb = tok >> 11, s = tok & (SEQ - 1);
                if (fbase < HIDDEN) {                      // Q, pre-scaled, [b,h,s,d]
                    int h = fbase >> 6, d0 = fbase & 63;
                    v8bf o;
#pragma unroll
                    for (int r = 0; r < 8; ++r) o[r] = (__bf16)(acc[mt][nt][r] * QK_SCALE);
                    *(v8bf*)&Qb[((size_t)(bb * HEADS + h) * SEQ + s) * HEAD_DIM + d0] = o;
                } else if (fbase < 2 * HIDDEN) {           // K, [b,h,s,d]
                    int nn = fbase - HIDDEN;
                    int h = nn >> 6, d0 = nn & 63;
                    v8bf o;
#pragma unroll
                    for (int r = 0; r < 8; ++r) o[r] = (__bf16)acc[mt][nt][r];
                    *(v8bf*)&Kb[((size_t)(bb * HEADS + h) * SEQ + s) * HEAD_DIM + d0] = o;
                } else {                                   // V transposed, [b,h,d,s]
                    int nn = fbase - 2 * HIDDEN;
                    int h = nn >> 6, d0 = nn & 63;
#pragma unroll
                    for (int r = 0; r < 8; ++r)
                        Vt[((size_t)(bb * HEADS + h) * HEAD_DIM + d0 + r) * SEQ + s] =
                            (__bf16)acc[mt][nt][r];
                }
            }
        }
    }
}

// ---------------------------------------------------------------------------
// Flash attention, transposed: 1 wave per block, 32 queries, 64-key steps.
//   S^T = K Q^T  (keys x queries)  -> per-query stats are lane-local + one permlanex16
//   P^T -> LDS as [q][key] via packed ds_store_b128
//   O^T = V^T P^T (dims x queries) -> packed b128 stores of 8 consecutive dims
// ---------------------------------------------------------------------------
__global__ __launch_bounds__(32)
void flash_attn(const __bf16* __restrict__ Qb, const __bf16* __restrict__ Kb,
                const __bf16* __restrict__ Vt, __bf16* __restrict__ Ob) {
    __shared__ __align__(16) __bf16 Pl[32 * 64];   // [query][key]

    const int lane = threadIdx.x & 31;
    const int l16  = lane & 15;
    const int half = lane >> 4;
    const int bh = blockIdx.y;                 // b*HEADS + h
    const int q0 = blockIdx.x * 32;
    const int b  = bh >> 4;
    const int h  = bh & (HEADS - 1);

    const __bf16* Qbase = Qb + (size_t)bh * SEQ * HEAD_DIM;
    const __bf16* Kbase = Kb + (size_t)bh * SEQ * HEAD_DIM;
    const __bf16* Vbase = Vt + (size_t)bh * HEAD_DIM * SEQ;

    // Q as B-fragments (lane = query, K = head dims), resident all pass
    v16bf qf[2][2];
#pragma unroll
    for (int nt = 0; nt < 2; ++nt)
#pragma unroll
        for (int ks = 0; ks < 2; ++ks)
            qf[nt][ks] = load_b_frag(Qbase + (size_t)(q0 + nt * 16 + l16) * HEAD_DIM,
                                     ks * 32 + half * 8);

    v8f oacc[4][2] = {};                 // O^T accumulators [dim tile][query tile]
    float mst[2] = {-3.0e38f, -3.0e38f}; // per-lane query stats (lane's query = nt*16+l16)
    float lst[2] = {0.f, 0.f};

    for (int kk0 = 0; kk0 < SEQ; kk0 += 64) {
        // ---- S^T = K Q^T ----
        v8f sacc[4][2] = {};
#pragma unroll
        for (int mtK = 0; mtK < 4; ++mtK) {
            v16bf kfa[2];
#pragma unroll
            for (int ks = 0; ks < 2; ++ks)
                kfa[ks] = load_a_frag(Kbase + (size_t)(kk0 + mtK * 16 + l16) * HEAD_DIM,
                                      ks * 32 + half * 8);
#pragma unroll
            for (int ntQ = 0; ntQ < 2; ++ntQ)
#pragma unroll
                for (int ks = 0; ks < 2; ++ks)
                    sacc[mtK][ntQ] = wmma_bf16(kfa[ks], qf[ntQ][ks], sacc[mtK][ntQ]);
        }

        // ---- online softmax; all reductions lane-local except one cross-half op ----
#pragma unroll
        for (int ntQ = 0; ntQ < 2; ++ntQ) {
            float mx = sacc[0][ntQ][0];
#pragma unroll
            for (int mtK = 0; mtK < 4; ++mtK)
#pragma unroll
                for (int r = 0; r < 8; ++r)
                    mx = fmaxf(mx, sacc[mtK][ntQ][r]);
            mx = fmaxf(mx, xhalf(mx));                 // combine the two key-halves
            float mnew = fmaxf(mst[ntQ], mx);
            float corr = __expf(mst[ntQ] - mnew);
            mst[ntQ] = mnew;

            float rsum = 0.f;
#pragma unroll
            for (int mtK = 0; mtK < 4; ++mtK) {
                v8bf pk;
#pragma unroll
                for (int r = 0; r < 8; ++r) {
                    float p = __expf(sacc[mtK][ntQ][r] - mnew);
                    rsum += p;
                    pk[r] = (__bf16)p;
                }
                // 8 consecutive keys for this lane's query: one packed 16B LDS store
                *(v8bf*)&Pl[(ntQ * 16 + l16) * 64 + mtK * 16 + half * 8] = pk;
            }
            rsum += xhalf(rsum);
            lst[ntQ] = lst[ntQ] * corr + rsum;
#pragma unroll
            for (int mtD = 0; mtD < 4; ++mtD)
                oacc[mtD][ntQ] *= corr;                // lane-scalar rescale of O^T
        }
        // Single wave: LDS ops are issued in order; the compiler inserts the
        // required s_wait_dscnt on the read-after-write below.

        // ---- O^T += V^T P^T ----
        v16bf pfb[2][2];
#pragma unroll
        for (int ntQ = 0; ntQ < 2; ++ntQ)
#pragma unroll
            for (int ks = 0; ks < 2; ++ks)
                pfb[ntQ][ks] = load_b_frag(&Pl[(ntQ * 16 + l16) * 64], ks * 32 + half * 8);
#pragma unroll
        for (int mtD = 0; mtD < 4; ++mtD) {
            v16bf vfa[2];
#pragma unroll
            for (int ks = 0; ks < 2; ++ks)
                vfa[ks] = load_a_frag(Vbase + (size_t)(mtD * 16 + l16) * SEQ + kk0,
                                      ks * 32 + half * 8);
#pragma unroll
            for (int ntQ = 0; ntQ < 2; ++ntQ)
#pragma unroll
                for (int ks = 0; ks < 2; ++ks)
                    oacc[mtD][ntQ] = wmma_bf16(vfa[ks], pfb[ntQ][ks], oacc[mtD][ntQ]);
        }
    }

    // ---- normalize, pack 8 consecutive dims per lane, store b128 ----
#pragma unroll
    for (int ntQ = 0; ntQ < 2; ++ntQ) {
        float inv = 1.0f / lst[ntQ];
        int q = q0 + ntQ * 16 + l16;
#pragma unroll
        for (int mtD = 0; mtD < 4; ++mtD) {
            v8bf o;
#pragma unroll
            for (int r = 0; r < 8; ++r) o[r] = (__bf16)(oacc[mtD][ntQ][r] * inv);
            *(v8bf*)&Ob[(size_t)(b * SEQ + q) * HIDDEN + h * HEAD_DIM + mtD * 16 + half * 8] = o;
        }
    }
}

// ---------------------------------------------------------------------------
extern "C" void kernel_launch(void* const* d_in, const int* in_sizes, int n_in,
                              void* d_out, int out_size, void* d_ws, size_t ws_size,
                              hipStream_t stream) {
    const float* x     = (const float*)d_in[0];   // [2,2048,1024]
    const float* qkv_w = (const float*)d_in[1];   // [3072,1024]
    const float* out_w = (const float*)d_in[2];   // [1024,1024]
    float* out = (float*)d_out;                   // [2,2048,1024]

    char* ws = (char*)d_ws;                       // needs 48 MB
    __bf16* xb   = (__bf16*)(ws);                            // 8 MB  [4096,1024]
    __bf16* wqkv = (__bf16*)(ws + (size_t)8  * 1024 * 1024); // 6 MB  [3072,1024]
    __bf16* wout = (__bf16*)(ws + (size_t)14 * 1024 * 1024); // 2 MB  [1024,1024]
    __bf16* Qb   = (__bf16*)(ws + (size_t)16 * 1024 * 1024); // 8 MB  [b,h,s,d]
    __bf16* Kb   = (__bf16*)(ws + (size_t)24 * 1024 * 1024); // 8 MB  [b,h,s,d]
    __bf16* Vt   = (__bf16*)(ws + (size_t)32 * 1024 * 1024); // 8 MB  [b,h,d,s]
    __bf16* Ob   = (__bf16*)(ws + (size_t)40 * 1024 * 1024); // 8 MB  [4096,1024]

    // fp32 -> bf16
    cvt_f32_bf16<<<(ROWS * HIDDEN / 4 + 255) / 256, 256, 0, stream>>>(x, xb, ROWS * HIDDEN);
    cvt_f32_bf16<<<(3 * HIDDEN * HIDDEN / 4 + 255) / 256, 256, 0, stream>>>(qkv_w, wqkv,
                                                                            3 * HIDDEN * HIDDEN);
    cvt_f32_bf16<<<(HIDDEN * HIDDEN / 4 + 255) / 256, 256, 0, stream>>>(out_w, wout,
                                                                        HIDDEN * HIDDEN);

    // QKV projection (transposed: A=weights, B=tokens) + per-head scatter
    dim3 g1(ROWS / 128, 3 * HIDDEN / 128);
    gemm_bf16<0><<<g1, 128, 0, stream>>>(wqkv, xb, HIDDEN, 0, nullptr, Qb, Kb, Vt);

    // Flash attention (transposed tiles)
    dim3 g2(SEQ / 32, BATCH * HEADS);
    flash_attn<<<g2, 32, 0, stream>>>(Qb, Kb, Vt, Ob);

    // Output projection (transposed), fp32 packed rows into d_out
    dim3 g3(ROWS / 128, HIDDEN / 128);
    gemm_bf16<1><<<g3, 128, 0, stream>>>(wout, Ob, HIDDEN, HIDDEN, out, nullptr, nullptr, nullptr);
}
